// SoftClusterGNN_21973052686420
// MI455X (gfx1250) — matrix operand order repaired and checked by reading, not compiled, not run on hardware
//
#include <hip/hip_runtime.h>
#include <hip/hip_bf16.h>

// ---------------- problem constants (match reference) ----------------
#define NB   8
#define NPGc 10000
#define NNc  80000
#define EEc  1280000
#define DINc 512
#define HIDc 128

typedef __attribute__((ext_vector_type(16))) unsigned short v16u;
typedef __attribute__((ext_vector_type(16))) __bf16        v16bf;
typedef __attribute__((ext_vector_type(8)))  float         v8f;

union BF16x16 { v16u u; v16bf b; };

// native RNE f32->bf16 (lowers to v_cvt_*bf16* on gfx1250 instead of a
// bfe/add3/mov_b16 emulation chain)
__device__ __forceinline__ __bf16 f2bf(float f) { return (__bf16)f; }
__device__ __forceinline__ unsigned short bfbits(float f) {
    __bf16 h = (__bf16)f;
    return __builtin_bit_cast(unsigned short, h);
}

// =====================================================================
// WMMA GEMM: xw0 = x(80000x512) @ W0(512x128), bf16 operands / f32 accum
// Block = 256 threads = 8 waves; each wave owns a 16-row strip, all 128 cols.
// W0 is staged per-128-K-slab into LDS, pre-swizzled into B-fragment order
// so each lane reads its 16 bf16 B elements as one contiguous 32B chunk
// (two ds_load_b128 per fragment).
// =====================================================================
__global__ __launch_bounds__(256) void k_gemm_xw0(const float* __restrict__ x,
                                                  const float* __restrict__ W,
                                                  float* __restrict__ xw) {
    __shared__ unsigned short sb[16384];                 // 32 KB: one 128xK slab
    const int tid  = threadIdx.x;
    const int wv   = tid >> 5;
    const int lane = tid & 31;
    const int hf   = lane >> 4;
    const int l16  = lane & 15;
    const int mrow = blockIdx.x * 128 + wv * 16 + l16;   // A row for this lane

    v8f acc[8] = {};

    for (int ks = 0; ks < DINc; ks += 128) {
        __syncthreads();
        // stage W[ks..ks+128)[0..128) as bf16 in B-fragment order
        for (int i = 0; i < 64; ++i) {
            int idx = i * 256 + tid;
            int k = idx >> 7, n = idx & 127;
            int ck = k >> 5, kk = k & 31;
            int tl = (((kk >> 3) & 1) << 4) | (n & 15);
            int te = (kk & 7) | ((kk >> 4) << 3);
            int tj = n >> 4;
            sb[((ck * 8 + tj) * 32 + tl) * 16 + te] = bfbits(W[(ks + k) * HIDc + n]);
        }
        __syncthreads();
        for (int ck = 0; ck < 4; ++ck) {
            // A fragment (16x32): two runs of 8 consecutive K per lane
            const float* xr = x + (size_t)mrow * DINc + ks + ck * 32 + hf * 8;
            float4 p0 = *(const float4*)(xr);
            float4 p1 = *(const float4*)(xr + 4);
            float4 q0 = *(const float4*)(xr + 16);
            float4 q1 = *(const float4*)(xr + 20);
            v16bf a;
            a[0]  = f2bf(p0.x); a[1]  = f2bf(p0.y); a[2]  = f2bf(p0.z); a[3]  = f2bf(p0.w);
            a[4]  = f2bf(p1.x); a[5]  = f2bf(p1.y); a[6]  = f2bf(p1.z); a[7]  = f2bf(p1.w);
            a[8]  = f2bf(q0.x); a[9]  = f2bf(q0.y); a[10] = f2bf(q0.z); a[11] = f2bf(q0.w);
            a[12] = f2bf(q1.x); a[13] = f2bf(q1.y); a[14] = f2bf(q1.z); a[15] = f2bf(q1.w);
            #pragma unroll
            for (int j = 0; j < 8; ++j) {
                BF16x16 bb;
                bb.u = *(const v16u*)&sb[((ck * 8 + j) * 32 + lane) * 16];
                acc[j] = __builtin_amdgcn_wmma_f32_16x16x32_bf16(
                    false, a, false, bb.b, (short)0, acc[j], false, false);
            }
        }
    }
    const int mb = blockIdx.x * 128 + wv * 16 + hf * 8;
    #pragma unroll
    for (int j = 0; j < 8; ++j)
        #pragma unroll
        for (int r = 0; r < 8; ++r)
            xw[(size_t)(mb + r) * HIDc + j * 16 + l16] = acc[j][r];
}

// =====================================================================
// WMMA pooled features: new_x[b] = probs0[b]^T (100x10000) @ h0[b] (10000x128)
// One block per graph; waves 0..6 own 16-cluster m-tiles (112 >= 100).
// =====================================================================
__global__ __launch_bounds__(256) void k_pool0(const float* __restrict__ probs,
                                               const float* __restrict__ h,
                                               float* __restrict__ nx) {
    const int b = blockIdx.x;
    const int wv = threadIdx.x >> 5;
    if (wv >= 7) return;
    const int lane = threadIdx.x & 31;
    const int hf = lane >> 4, l16 = lane & 15;
    const int m = wv * 16 + l16;
    const size_t pb = (size_t)b * NPGc;
    v8f acc[8] = {};

    for (int kc = 0; kc < NPGc; kc += 32) {
        v16bf a;
        #pragma unroll
        for (int e = 0; e < 16; ++e) {
            int k = kc + hf * 16 + e;
            float v = (k < NPGc && m < 100) ? probs[(pb + k) * 100 + m] : 0.0f;
            a[e] = f2bf(v);
        }
        #pragma unroll
        for (int j = 0; j < 8; ++j) {
            int n = j * 16 + l16;
            v16bf bb;
            #pragma unroll
            for (int e = 0; e < 16; ++e) {
                int k = kc + hf * 16 + e;
                float v = (k < NPGc) ? h[(pb + k) * HIDc + n] : 0.0f;
                bb[e] = f2bf(v);
            }
            acc[j] = __builtin_amdgcn_wmma_f32_16x16x32_bf16(
                false, a, false, bb, (short)0, acc[j], false, false);
        }
    }
    const int mb = wv * 16 + hf * 8;
    for (int j = 0; j < 8; ++j)
        for (int r = 0; r < 8; ++r) {
            int mm = mb + r;
            if (mm < 100)
                nx[((size_t)b * 100 + mm) * HIDc + j * 16 + l16] = acc[j][r];
        }
}

// ---------------- scalar / atomic helper kernels ----------------
__global__ void k_fill(float* p, float v, long n) {
    long i = (long)blockIdx.x * 256 + threadIdx.x;
    if (i < n) p[i] = v;
}
__global__ void k_deg(const int* __restrict__ dst, float* deg, long E) {
    long e = (long)blockIdx.x * 256 + threadIdx.x;
    if (e < E) atomicAdd(&deg[dst[e]], 1.0f);
}
__global__ void k_dinv(float* d, long n) {
    long i = (long)blockIdx.x * 256 + threadIdx.x;
    if (i < n) d[i] = rsqrtf(d[i] + 1.0f);
}
// out = xw * dinv^2 + bias   (self-loop term of GCN)
__global__ void k_self(const float* __restrict__ xw, const float* __restrict__ dinv,
                       const float* __restrict__ bias, float* __restrict__ out,
                       long n, int F) {
    long i = (long)blockIdx.x * 256 + threadIdx.x;
    if (i >= n * F) return;
    long r = i / F; int f = i % F;
    float di = dinv[r];
    out[i] = xw[i] * di * di + bias[f];
}
// out[dst] += xw[src]*dinv[src]*dinv[dst]
__global__ void k_edge_agg(const float* __restrict__ xw, const float* __restrict__ dinv,
                           const int* __restrict__ src, const int* __restrict__ dst,
                           float* __restrict__ out, long E, int F) {
    long i = (long)blockIdx.x * 256 + threadIdx.x;
    if (i >= E * F) return;
    long e = i / F; int f = i % F;
    int s = src[e], d = dst[e];
    atomicAdd(&out[(size_t)d * F + f], xw[(size_t)s * F + f] * dinv[s] * dinv[d]);
}
__global__ void k_relu(float* p, long n) {
    long i = (long)blockIdx.x * 256 + threadIdx.x;
    if (i < n) p[i] = fmaxf(p[i], 0.0f);
}
// per-node l2norm of 2d coord then @ W1(2 x hd)
__global__ void k_l2w(const float* __restrict__ coord, const float* __restrict__ W1,
                      float* __restrict__ out, long n, int hd) {
    long i = (long)blockIdx.x * 256 + threadIdx.x;
    if (i >= n) return;
    float x0 = coord[i * 2], x1 = coord[i * 2 + 1];
    float inv = 1.0f / fmaxf(sqrtf(x0 * x0 + x1 * x1), 1e-12f);
    x0 *= inv; x1 *= inv;
    for (int j = 0; j < hd; ++j) out[i * hd + j] = x0 * W1[j] + x1 * W1[hd + j];
}
__global__ void k_l2n(const float* __restrict__ in, float* __restrict__ out, long n) {
    long i = (long)blockIdx.x * 256 + threadIdx.x;
    if (i >= n) return;
    float x0 = in[i * 2], x1 = in[i * 2 + 1];
    float inv = 1.0f / fmaxf(sqrtf(x0 * x0 + x1 * x1), 1e-12f);
    out[i * 2] = x0 * inv; out[i * 2 + 1] = x1 * inv;
}
// BN stats over npg nodes per (graph,feature): block-per-(g,f) reduction
__global__ void k_bn_stats(const float* __restrict__ x, float* mean, float* var,
                           int npg, int F) {
    __shared__ float s1[256], s2[256];
    int f = blockIdx.x % F, g = blockIdx.x / F;
    float a = 0.f, b = 0.f;
    for (int p = threadIdx.x; p < npg; p += 256) {
        float v = x[((size_t)g * npg + p) * F + f];
        a += v; b += v * v;
    }
    s1[threadIdx.x] = a; s2[threadIdx.x] = b;
    __syncthreads();
    for (int s = 128; s > 0; s >>= 1) {
        if ((int)threadIdx.x < s) { s1[threadIdx.x] += s1[threadIdx.x + s]; s2[threadIdx.x] += s2[threadIdx.x + s]; }
        __syncthreads();
    }
    if (threadIdx.x == 0) {
        float m = s1[0] / npg;
        mean[blockIdx.x] = m;
        var[blockIdx.x] = s2[0] / npg - m * m;
    }
}
// small serial BN stats (dense levels, P<=100)
__global__ void k_bn_small(const float* __restrict__ x, float* mean, float* var,
                           int P, int F, int total) {
    int i = blockIdx.x * 256 + threadIdx.x;
    if (i >= total) return;
    int f = i % F, b = i / F;
    float s = 0.f, q = 0.f;
    for (int p = 0; p < P; ++p) { float v = x[((size_t)b * P + p) * F + f]; s += v; q += v * v; }
    float m = s / P; mean[i] = m; var[i] = q / P - m * m;
}
__global__ void k_bn_relu(float* __restrict__ x, const float* __restrict__ mean,
                          const float* __restrict__ var, const float* __restrict__ gam,
                          const float* __restrict__ bet, int npg, int F, long n) {
    long i = (long)blockIdx.x * 256 + threadIdx.x;
    if (i >= n) return;
    int f = (int)(i % F);
    int g = (int)(i / ((long)npg * F));
    float y = (x[i] - mean[g * F + f]) * rsqrtf(var[g * F + f] + 1e-5f) * gam[f] + bet[f];
    x[i] = fmaxf(y, 0.0f);
}
__global__ void k_matmul(const float* __restrict__ xin, const float* __restrict__ W,
                         float* __restrict__ out, long rows, int fin, int fout) {
    long i = (long)blockIdx.x * 256 + threadIdx.x;
    if (i >= rows * fout) return;
    int fo = (int)(i % fout); long r = i / fout;
    const float* xr = xin + r * fin;
    float acc = 0.f;
    for (int k = 0; k < fin; ++k) acc += xr[k] * W[k * fout + fo];
    out[i] = acc;
}
// logits = h@Wo + bo, softmax over C (block per row, 128 threads)
__global__ void k_softmax_head(const float* __restrict__ h, const float* __restrict__ Wo,
                               const float* __restrict__ bo, float* __restrict__ probs,
                               int hd, int C) {
    __shared__ float s[128];
    long node = blockIdx.x;
    int t = threadIdx.x;
    float logit = -3.4e38f;
    if (t < C) {
        const float* hr = h + node * hd;
        float a = bo[t];
        for (int k = 0; k < hd; ++k) a += hr[k] * Wo[k * C + t];
        logit = a;
    }
    s[t] = logit; __syncthreads();
    for (int st = 64; st > 0; st >>= 1) { if (t < st) s[t] = fmaxf(s[t], s[t + st]); __syncthreads(); }
    float mx = s[0]; __syncthreads();
    float e = (t < C) ? __expf(logit - mx) : 0.0f;
    s[t] = e; __syncthreads();
    for (int st = 64; st > 0; st >>= 1) { if (t < st) s[t] += s[t + st]; __syncthreads(); }
    if (t < C) probs[node * C + t] = e / s[0];
}
__global__ void k_argmax(const float* __restrict__ probs, int* __restrict__ cl, long n, int C) {
    long i = (long)blockIdx.x * 256 + threadIdx.x;
    if (i >= n) return;
    const float* pr = probs + i * C;
    int best = 0; float bv = pr[0];
    for (int c = 1; c < C; ++c) { float v = pr[c]; if (v > bv) { bv = v; best = c; } }
    cl[i] = best;
}
__global__ void k_cnt0(const int* __restrict__ src, const int* __restrict__ dst,
                       const int* __restrict__ cl, float* __restrict__ cnt,
                       long E, int npg, int C) {
    long e = (long)blockIdx.x * 256 + threadIdx.x;
    if (e >= E) return;
    int s = src[e], d = dst[e];
    int b = s / npg;
    atomicAdd(&cnt[((size_t)b * C + cl[s]) * C + cl[d]], 1.0f);
}
__global__ void k_adj_fin(float* __restrict__ A, int C, long total) {
    long i = (long)blockIdx.x * 256 + threadIdx.x;
    if (i >= total) return;
    int col = (int)(i % C);
    int row = (int)((i / C) % C);
    A[i] = (A[i] > 0.0f && row != col) ? 1.0f : 0.0f;
}
__global__ void k_coord_pool0(const float* __restrict__ crd, const int* __restrict__ cl,
                              float* __restrict__ csum, float* __restrict__ ccnt,
                              long n, int npg, int C) {
    long i = (long)blockIdx.x * 256 + threadIdx.x;
    if (i >= n) return;
    int b = (int)(i / npg);
    int c = cl[i];
    atomicAdd(&csum[((size_t)b * C + c) * 2], crd[i * 2]);
    atomicAdd(&csum[((size_t)b * C + c) * 2 + 1], crd[i * 2 + 1]);
    atomicAdd(&ccnt[b * C + c], 1.0f);
}
__global__ void k_coord_fin(const float* __restrict__ csum, const float* __restrict__ ccnt,
                            float* __restrict__ out, long n) {
    long i = (long)blockIdx.x * 256 + threadIdx.x;
    if (i >= n) return;
    float c = ccnt[i];
    if (c > 0.f) { out[i * 2] = csum[i * 2] / c; out[i * 2 + 1] = csum[i * 2 + 1] / c; }
    else         { out[i * 2] = 0.f; out[i * 2 + 1] = 0.f; }
}
// ---- dense-level GCN helpers (P <= 100) ----
__global__ void k_dense_dinv(const float* __restrict__ A, float* __restrict__ dinv,
                             int P, int total) {
    int i = blockIdx.x * 256 + threadIdx.x;
    if (i >= total) return;
    int b = i / P, d = i % P;
    float s = 0.f;
    for (int r = 0; r < P; ++r) s += A[((size_t)b * P + r) * P + d];
    dinv[i] = rsqrtf(s + 1.0f);
}
__global__ void k_dense_agg(const float* __restrict__ xw, const float* __restrict__ A,
                            const float* __restrict__ dinv, const float* __restrict__ bias,
                            float* __restrict__ out, int P, int F, int relu, long total) {
    long i = (long)blockIdx.x * 256 + threadIdx.x;
    if (i >= total) return;
    int f = (int)(i % F); long t = i / F;
    int d = (int)(t % P); int b = (int)(t / P);
    const float* Ab = A + (size_t)b * P * P;
    const float* dv = dinv + b * P;
    float acc = 0.f;
    for (int s = 0; s < P; ++s)
        acc += Ab[(size_t)s * P + d] * dv[s] * xw[((size_t)b * P + s) * F + f];
    float dd = dv[d];
    float y = acc * dd + xw[((size_t)b * P + d) * F + f] * dd * dd + bias[f];
    out[i] = relu ? fmaxf(y, 0.0f) : y;
}
__global__ void k_pool_x(const float* __restrict__ probs, const float* __restrict__ h,
                         float* __restrict__ nx, int P, int C, int F, long total) {
    long i = (long)blockIdx.x * 256 + threadIdx.x;
    if (i >= total) return;
    int f = (int)(i % F); long t = i / F;
    int c = (int)(t % C); int b = (int)(t / C);
    float acc = 0.f;
    for (int p = 0; p < P; ++p)
        acc += probs[((size_t)b * P + p) * C + c] * h[((size_t)b * P + p) * F + f];
    nx[i] = acc;
}
__global__ void k_cnt_dense(const float* __restrict__ A, const int* __restrict__ cl,
                            float* __restrict__ cnt, int P, int C, long total) {
    long i = (long)blockIdx.x * 256 + threadIdx.x;
    if (i >= total) return;
    int q = (int)(i % P); long t = i / P;
    int p = (int)(t % P); int b = (int)(t / P);
    if (A[i] != 0.0f)
        atomicAdd(&cnt[((size_t)b * C + cl[b * P + p]) * C + cl[b * P + q]], 1.0f);
}
__global__ void k_coord_pool_d(const float* __restrict__ crd, const int* __restrict__ cl,
                               float* __restrict__ csum, float* __restrict__ ccnt,
                               int P, int C, long n) {
    long i = (long)blockIdx.x * 256 + threadIdx.x;
    if (i >= n) return;
    int b = (int)(i / P);
    int c = cl[i];
    atomicAdd(&csum[((size_t)b * C + c) * 2], crd[i * 2]);
    atomicAdd(&csum[((size_t)b * C + c) * 2 + 1], crd[i * 2 + 1]);
    atomicAdd(&ccnt[b * C + c], 1.0f);
}
__global__ void k_mean(const float* __restrict__ x, float* __restrict__ out,
                       int P, int F, long n) {
    long i = (long)blockIdx.x * 256 + threadIdx.x;
    if (i >= n) return;
    int b = (int)(i / F), f = (int)(i % F);
    float s = 0.f;
    for (int p = 0; p < P; ++p) s += x[((size_t)b * P + p) * F + f];
    out[i] = s / P;
}

static inline unsigned gs(long n) { return (unsigned)((n + 255) / 256); }

extern "C" void kernel_launch(void* const* d_in, const int* in_sizes, int n_in,
                              void* d_out, int out_size, void* d_ws, size_t ws_size,
                              hipStream_t stream) {
    const float* x     = (const float*)d_in[0];
    const float* coord = (const float*)d_in[1];
    const int*   eidx  = (const int*)d_in[2];
    const long EE = in_sizes[2] / 2;
    const int* src = eidx;
    const int* dst = eidx + EE;
    const float* cW0 = (const float*)d_in[4],  *cB0 = (const float*)d_in[5];
    const float* cW1 = (const float*)d_in[6],  *cB1 = (const float*)d_in[7];
    const float* cW2 = (const float*)d_in[8],  *cB2 = (const float*)d_in[9];
    const float* cW3 = (const float*)d_in[10], *cB3 = (const float*)d_in[11];
    const float* P0[10]; const float* P1[10]; const float* P2[10];
    for (int i = 0; i < 10; ++i) { P0[i] = (const float*)d_in[12 + i];
                                   P1[i] = (const float*)d_in[22 + i];
                                   P2[i] = (const float*)d_in[32 + i]; }

    // ---- workspace carve ----
    char* w = (char*)d_ws; size_t off = 0;
    auto mk = [&](size_t elems) -> float* {
        float* p = (float*)(w + off);
        off += ((elems * 4 + 255) / 256) * 256;
        return p;
    };
    float* dinv  = mk(NNc);
    float* xw0   = mk((size_t)NNc * HIDc);        // reused later as pred scratch
    float* h0    = mk((size_t)NNc * HIDc);
    float* pr0   = mk((size_t)NNc * 100);
    int*   cl0   = (int*)mk(NNc);
    float* mean  = mk(8 * 128);
    float* var   = mk(8 * 128);
    float* nxA   = mk(8 * 100 * 128);
    float* nxB   = mk(8 * 100 * 128);
    float* Aa    = mk(8 * 100 * 100);
    float* Ab_   = mk(8 * 100 * 100);
    float* crdA  = mk(8 * 100 * 2);
    float* crdB  = mk(8 * 100 * 2);
    float* csum  = mk(8 * 100 * 2);
    float* ccnt  = mk(8 * 100);
    int*   cld   = (int*)mk(8 * 100);
    float* dinvd = mk(8 * 100);
    float* dxw   = mk(8 * 100 * 128);
    float* dpr   = mk(8 * 100 * 100);
    float* dhA   = mk(8 * 100 * 32);
    float* dhB   = mk(8 * 100 * 32);
    float* dcn   = mk(8 * 100 * 2);
    // pred-sparse scratch aliases xw0 region (xw0 dead after h0 is built)
    float* hpXW = xw0;
    float* hpA  = xw0 + (size_t)NNc * 50;

    // ---- degrees / dinv ----
    k_fill<<<gs(NNc), 256, 0, stream>>>(dinv, 0.f, NNc);
    k_deg <<<gs(EE), 256, 0, stream>>>(dst, dinv, EE);
    k_dinv<<<gs(NNc), 256, 0, stream>>>(dinv, NNc);

    // ---- level-0 sparse GCN: h0 = relu(agg(x@W0) + self + b0) ----
    k_gemm_xw0<<<dim3(NNc / 128), dim3(256), 0, stream>>>(x, cW0, xw0);
    k_self    <<<gs((long)NNc * HIDc), 256, 0, stream>>>(xw0, dinv, cB0, h0, NNc, HIDc);
    k_edge_agg<<<gs(EE * HIDc), 256, 0, stream>>>(xw0, dinv, src, dst, h0, EE, HIDc);
    k_relu    <<<gs((long)NNc * HIDc), 256, 0, stream>>>(h0, (long)NNc * HIDc);

    // ---- level-0 sparse prediction head (hd=50, C=100), all graphs fused ----
    k_l2w     <<<gs(NNc), 256, 0, stream>>>(coord, P0[0], hpXW, NNc, 50);
    k_self    <<<gs((long)NNc * 50), 256, 0, stream>>>(hpXW, dinv, P0[1], hpA, NNc, 50);
    k_edge_agg<<<gs(EE * 50), 256, 0, stream>>>(hpXW, dinv, src, dst, hpA, EE, 50);
    k_bn_stats<<<dim3(8 * 50), dim3(256), 0, stream>>>(hpA, mean, var, NPGc, 50);
    k_bn_relu <<<gs((long)NNc * 50), 256, 0, stream>>>(hpA, mean, var, P0[2], P0[3], NPGc, 50, (long)NNc * 50);
    k_matmul  <<<gs((long)NNc * 50), 256, 0, stream>>>(hpA, P0[4], hpXW, NNc, 50, 50);
    k_self    <<<gs((long)NNc * 50), 256, 0, stream>>>(hpXW, dinv, P0[5], hpA, NNc, 50);
    k_edge_agg<<<gs(EE * 50), 256, 0, stream>>>(hpXW, dinv, src, dst, hpA, EE, 50);
    k_bn_stats<<<dim3(8 * 50), dim3(256), 0, stream>>>(hpA, mean, var, NPGc, 50);
    k_bn_relu <<<gs((long)NNc * 50), 256, 0, stream>>>(hpA, mean, var, P0[6], P0[7], NPGc, 50, (long)NNc * 50);
    k_softmax_head<<<dim3(NNc), dim3(128), 0, stream>>>(hpA, P0[8], P0[9], pr0, 50, 100);

    // ---- pool level 0 ----
    k_pool0 <<<dim3(NB), dim3(256), 0, stream>>>(pr0, h0, nxA);
    k_argmax<<<gs(NNc), 256, 0, stream>>>(pr0, cl0, NNc, 100);
    k_fill  <<<gs(8 * 100 * 100), 256, 0, stream>>>(Aa, 0.f, 8 * 100 * 100);
    k_cnt0  <<<gs(EE), 256, 0, stream>>>(src, dst, cl0, Aa, EE, NPGc, 100);
    k_adj_fin<<<gs(8 * 100 * 100), 256, 0, stream>>>(Aa, 100, 8 * 100 * 100);
    k_fill  <<<gs(8 * 100 * 2), 256, 0, stream>>>(csum, 0.f, 8 * 100 * 2);
    k_fill  <<<gs(8 * 100), 256, 0, stream>>>(ccnt, 0.f, 8 * 100);
    k_coord_pool0<<<gs(NNc), 256, 0, stream>>>(coord, cl0, csum, ccnt, NNc, NPGc, 100);
    k_coord_fin  <<<gs(8 * 100), 256, 0, stream>>>(csum, ccnt, crdA, 8 * 100);

    // ---- dense levels 1 and 2 ----
    float* h  = nxA;  float* hn = nxB;
    float* Ac = Aa;   float* An = Ab_;
    float* cc = crdA; float* cn = crdB;
    int P = 100;
    const float* convW[2] = {cW1, cW2};
    const float* convB[2] = {cB1, cB2};
    const float** preds[2] = {P1, P2};
    const int CSn[2] = {50, 10};
    const int HDn[2] = {25, 5};
    for (int lv = 0; lv < 2; ++lv) {
        int C = CSn[lv], hd = HDn[lv];
        const float** pp = preds[lv];
        // conv: h = relu(gcn_dense(h, A))
        k_dense_dinv<<<gs(8 * P), 256, 0, stream>>>(Ac, dinvd, P, 8 * P);
        k_matmul    <<<gs((long)8 * P * 128), 256, 0, stream>>>(h, convW[lv], dxw, 8L * P, 128, 128);
        k_dense_agg <<<gs((long)8 * P * 128), 256, 0, stream>>>(dxw, Ac, dinvd, convB[lv], h, P, 128, 1, (long)8 * P * 128);
        // pred on coords
        k_l2n       <<<gs(8 * P), 256, 0, stream>>>(cc, dcn, 8 * P);
        k_matmul    <<<gs((long)8 * P * hd), 256, 0, stream>>>(dcn, pp[0], dhA, 8L * P, 2, hd);
        k_dense_agg <<<gs((long)8 * P * hd), 256, 0, stream>>>(dhA, Ac, dinvd, pp[1], dhB, P, hd, 0, (long)8 * P * hd);
        k_bn_small  <<<gs(8 * hd), 256, 0, stream>>>(dhB, mean, var, P, hd, 8 * hd);
        k_bn_relu   <<<gs((long)8 * P * hd), 256, 0, stream>>>(dhB, mean, var, pp[2], pp[3], P, hd, (long)8 * P * hd);
        k_matmul    <<<gs((long)8 * P * hd), 256, 0, stream>>>(dhB, pp[4], dhA, 8L * P, hd, hd);
        k_dense_agg <<<gs((long)8 * P * hd), 256, 0, stream>>>(dhA, Ac, dinvd, pp[5], dhB, P, hd, 0, (long)8 * P * hd);
        k_bn_small  <<<gs(8 * hd), 256, 0, stream>>>(dhB, mean, var, P, hd, 8 * hd);
        k_bn_relu   <<<gs((long)8 * P * hd), 256, 0, stream>>>(dhB, mean, var, pp[6], pp[7], P, hd, (long)8 * P * hd);
        k_softmax_head<<<dim3(8 * P), dim3(128), 0, stream>>>(dhB, pp[8], pp[9], dpr, hd, C);
        // pool
        k_pool_x   <<<gs((long)8 * C * 128), 256, 0, stream>>>(dpr, h, hn, P, C, 128, (long)8 * C * 128);
        k_argmax   <<<gs(8 * P), 256, 0, stream>>>(dpr, cld, 8 * P, C);
        k_fill     <<<gs((long)8 * C * C), 256, 0, stream>>>(An, 0.f, (long)8 * C * C);
        k_cnt_dense<<<gs((long)8 * P * P), 256, 0, stream>>>(Ac, cld, An, P, C, (long)8 * P * P);
        k_adj_fin  <<<gs((long)8 * C * C), 256, 0, stream>>>(An, C, (long)8 * C * C);
        k_fill     <<<gs(8 * C * 2), 256, 0, stream>>>(csum, 0.f, 8 * C * 2);
        k_fill     <<<gs(8 * C), 256, 0, stream>>>(ccnt, 0.f, 8 * C);
        k_coord_pool_d<<<gs(8 * P), 256, 0, stream>>>(cc, cld, csum, ccnt, P, C, 8 * P);
        k_coord_fin<<<gs(8 * C), 256, 0, stream>>>(csum, ccnt, cn, 8 * C);
        // swap buffers
        float* t;
        t = h;  h  = hn; hn = t;
        t = Ac; Ac = An; An = t;
        t = cc; cc = cn; cn = t;
        P = C;
    }

    // ---- final conv (no relu) + global mean pool ----
    k_dense_dinv<<<gs(8 * P), 256, 0, stream>>>(Ac, dinvd, P, 8 * P);
    k_matmul    <<<gs((long)8 * P * 128), 256, 0, stream>>>(h, cW3, dxw, 8L * P, 128, 128);
    k_dense_agg <<<gs((long)8 * P * 128), 256, 0, stream>>>(dxw, Ac, dinvd, cB3, hn, P, 128, 0, (long)8 * P * 128);
    k_mean      <<<gs(8 * 128), 256, 0, stream>>>(hn, (float*)d_out, P, 128, 8 * 128);
}
